// Encoder_51805895524890
// MI455X (gfx1250) — matrix operand order repaired and checked by reading, not compiled, not run on hardware
//
#include <hip/hip_runtime.h>
#include <hip/hip_bf16.h>
#include <math.h>

// ---------------------------------------------------------------------------
// Transformer encoder forward (6 layers) for MI455X / gfx1250.
// GEMMs: v_wmma_f32_16x16x32_bf16 with DOUBLE-BUFFERED global_load_async_to_lds
// staging (copy of tile k+1 overlaps WMMA on tile k; one barrier per K-step).
// All GEMM operands live in bf16 (converted once by producers).
// ---------------------------------------------------------------------------

typedef __attribute__((ext_vector_type(16))) __bf16 v16bf;
typedef __attribute__((ext_vector_type(8)))  float  v8f;

#define D_MODEL 1024
#define NHEAD   16
#define DHEAD   64
#define DFF_    4096
#define NLAYER  6
#define BATCH   4
#define SEQ     512
#define ROWS    (BATCH * SEQ)      // 2048

#define BM 128
#define BN 64
#define BK 32
#define LDK (BK + 8)               // LDS row stride (bf16 elems), keeps 16B align
#define GT 256                     // 8 waves of 32

static __device__ __forceinline__ unsigned short f2bf(float f) {
  // round-to-nearest-even fp32 -> bf16
  unsigned int u = __float_as_uint(f);
  u += 0x7FFFu + ((u >> 16) & 1u);
  return (unsigned short)(u >> 16);
}

union Frag32B {               // 32 bytes: 16 bf16 lane-private fragment
  uint4 u[2];
  v16bf v;
};

static __device__ __forceinline__ unsigned ldsAddr(const void* p) {
  // low 32 bits of a flat shared pointer == LDS byte address
  return (unsigned)(unsigned long long)p;
}

static __device__ __forceinline__ void asyncCopy16B(unsigned lds,
                                                    const unsigned short* g) {
  asm volatile("global_load_async_to_lds_b128 %0, %1, off"
               :: "v"(lds), "v"(g) : "memory");
}

static __device__ __forceinline__ void waitAsync0() {
  asm volatile("s_wait_asynccnt 0" ::: "memory");
}

// MODE 0: plain GEMM, grid.z == 1
// MODE 1: scores  per z=(b*16+h): Cf = scale * Q Kt (+ key-pad mask -> -inf)
// MODE 2: AV      per z=(b*16+h): Cb = attn * V
// All B operands are [N,K] row-major (N-major).
template <int MODE>
__global__ __launch_bounds__(GT)
void wmma_gemm_bf16(const unsigned short* __restrict__ Abase, int lda,
                    const unsigned short* __restrict__ Btbase, int ldb,
                    float* __restrict__ Cf, int ldc,
                    unsigned short* __restrict__ Cb, int ldcb, int cbTrans,
                    const float* __restrict__ bias, int relu,
                    float scale, const int* __restrict__ tokens, int K)
{
  __shared__ __align__(16) unsigned short As[2][BM][LDK];  // double buffered
  __shared__ __align__(16) unsigned short Bs[2][BN][LDK];  // stored [n][k]

  const unsigned short* A  = Abase;
  const unsigned short* Bt = Btbase;
  const int* tokRow = nullptr;

  if (MODE == 1) {
    int z = blockIdx.z, b = z >> 4, h = z & 15;
    A  = Abase  + (size_t)(b * SEQ) * lda + h * DHEAD;   // Q slice [S, DH]
    Bt = Btbase + (size_t)(b * SEQ) * ldb + h * DHEAD;   // K slice [S, DH] (N-major)
    Cf = Cf + (size_t)z * SEQ * SEQ;                     // scores [S, S]
    tokRow = tokens + b * SEQ;
  } else if (MODE == 2) {
    int z = blockIdx.z, b = z >> 4, h = z & 15;
    A  = Abase  + (size_t)z * SEQ * SEQ;                 // attn [S, S] (bf16)
    Bt = Btbase + (size_t)z * DHEAD * SEQ;               // Vt slice [DH, S]
    Cb = Cb + (size_t)(b * SEQ) * ldcb + h * DHEAD;
  }

  const int t     = threadIdx.x;
  const int lane  = t & 31;
  const int wid   = t >> 5;
  const int waveM = wid & 3;        // 4 waves along M
  const int waveN = wid >> 2;       // 2 waves along N
  const int lg    = lane >> 4;      // lane group (0/1)
  const int lm    = lane & 15;
  const int rowBase = blockIdx.y * BM;
  const int colBase = blockIdx.x * BN;

  // ---- per-thread staging assignments (16B chunks), precomputed ----
  const int mA  = t >> 2;           // rows 0..63 (A chunk 0), +64 for chunk 1
  const int kcA = (t & 3) * 8;      // 4 chunks per 32-elem row
  const unsigned short* gA0 = A  + (size_t)(rowBase + mA)      * lda + kcA;
  const unsigned short* gA1 = A  + (size_t)(rowBase + 64 + mA) * lda + kcA;
  const unsigned short* gB  = Bt + (size_t)(colBase + mA)      * ldb + kcA;
  const unsigned ldsA0[2] = { ldsAddr(&As[0][mA][kcA]),      ldsAddr(&As[1][mA][kcA]) };
  const unsigned ldsA1[2] = { ldsAddr(&As[0][64 + mA][kcA]), ldsAddr(&As[1][64 + mA][kcA]) };
  const unsigned ldsB [2] = { ldsAddr(&Bs[0][mA][kcA]),      ldsAddr(&Bs[1][mA][kcA]) };

  const v8f vzero = {0.f, 0.f, 0.f, 0.f, 0.f, 0.f, 0.f, 0.f};
  v8f acc[2][2];
  acc[0][0] = vzero; acc[0][1] = vzero; acc[1][0] = vzero; acc[1][1] = vzero;

  const int nt = K / BK;

  // prologue: stage tile 0 into buffer 0
  asyncCopy16B(ldsA0[0], gA0);
  asyncCopy16B(ldsA1[0], gA1);
  asyncCopy16B(ldsB[0],  gB);
  gA0 += BK; gA1 += BK; gB += BK;

  for (int kt = 0; kt < nt; ++kt) {
    const int buf = kt & 1;
    waitAsync0();          // only tile kt's 3 async ops are outstanding
    __syncthreads();       // publishes tile kt; retires reads of buf^1 (tile kt-1)

    if (kt + 1 < nt) {     // stage tile kt+1 while computing tile kt
      const int nb = buf ^ 1;
      asyncCopy16B(ldsA0[nb], gA0);
      asyncCopy16B(ldsA1[nb], gA1);
      asyncCopy16B(ldsB[nb],  gB);
      gA0 += BK; gA1 += BK; gB += BK;
    }

    // ---- fragments per ISA 7.12.2 lane layouts ----
    Frag32B af[2], bfr[2];
#pragma unroll
    for (int i = 0; i < 2; ++i) {
      int row = waveM * 32 + i * 16 + lm;               // A: lane = row M
      af[i].u[0] = *reinterpret_cast<const uint4*>(&As[buf][row][8 * lg]);
      af[i].u[1] = *reinterpret_cast<const uint4*>(&As[buf][row][16 + 8 * lg]);
    }
#pragma unroll
    for (int j = 0; j < 2; ++j) {
      int col = waveN * 32 + j * 16 + lm;               // B: lane = col N
      bfr[j].u[0] = *reinterpret_cast<const uint4*>(&Bs[buf][col][8 * lg]);
      bfr[j].u[1] = *reinterpret_cast<const uint4*>(&Bs[buf][col][16 + 8 * lg]);
    }
#pragma unroll
    for (int i = 0; i < 2; ++i)
#pragma unroll
      for (int j = 0; j < 2; ++j)
        acc[i][j] = __builtin_amdgcn_wmma_f32_16x16x32_bf16(
            false, af[i].v, false, bfr[j].v, (short)0, acc[i][j], false, false);
  }

  // ---- epilogue (C layout: VGPR r -> M=r(+8 for hi lanes); N=lane%16) ----
  if (MODE == 1) {
#pragma unroll
    for (int i = 0; i < 2; ++i)
#pragma unroll
      for (int j = 0; j < 2; ++j) {
        int nG = colBase + waveN * 32 + j * 16 + lm;
        bool masked = (tokRow[nG] == 0);
#pragma unroll
        for (int r = 0; r < 8; ++r) {
          int mG = rowBase + waveM * 32 + i * 16 + r + 8 * lg;
          float val = acc[i][j][r] * scale;
          Cf[(size_t)mG * ldc + nG] = masked ? -INFINITY : val;
        }
      }
  } else {
    // fold bias (+relu) into acc
#pragma unroll
    for (int i = 0; i < 2; ++i)
#pragma unroll
      for (int j = 0; j < 2; ++j) {
        int nG = colBase + waveN * 32 + j * 16 + lm;
        float bv = bias ? bias[nG] : 0.0f;
#pragma unroll
        for (int r = 0; r < 8; ++r) {
          float val = acc[i][j][r] + bv;
          acc[i][j][r] = relu ? fmaxf(val, 0.0f) : val;
        }
      }
    if (Cf) {
#pragma unroll
      for (int i = 0; i < 2; ++i)
#pragma unroll
        for (int j = 0; j < 2; ++j) {
          int nG = colBase + waveN * 32 + j * 16 + lm;
#pragma unroll
          for (int r = 0; r < 8; ++r) {
            int mG = rowBase + waveM * 32 + i * 16 + r + 8 * lg;
            Cf[(size_t)mG * ldc + nG] = acc[i][j][r];
          }
        }
    }
    if (Cb) {
      if (cbTrans) {   // V projection: store [b,h,d,s] for AV's B operand
#pragma unroll
        for (int i = 0; i < 2; ++i)
#pragma unroll
          for (int j = 0; j < 2; ++j) {
            int nG = colBase + waveN * 32 + j * 16 + lm;
            int h = nG >> 6, d = nG & 63;
#pragma unroll
            for (int r = 0; r < 8; ++r) {
              int mG = rowBase + waveM * 32 + i * 16 + r + 8 * lg;
              int b = mG >> 9, s = mG & 511;
              Cb[(((size_t)b * NHEAD + h) * DHEAD + d) * SEQ + s] = f2bf(acc[i][j][r]);
            }
          }
      } else {
#pragma unroll
        for (int i = 0; i < 2; ++i)
#pragma unroll
          for (int j = 0; j < 2; ++j) {
            int nG = colBase + waveN * 32 + j * 16 + lm;
#pragma unroll
            for (int r = 0; r < 8; ++r) {
              int mG = rowBase + waveM * 32 + i * 16 + r + 8 * lg;
              Cb[(size_t)mG * ldcb + nG] = f2bf(acc[i][j][r]);
            }
          }
      }
    }
  }
}

// Weight convert+transpose: Wt[n][k] = bf16(W[k][n]); grid.z = layer
__global__ __launch_bounds__(256)
void conv_transpose_bf16(const float* __restrict__ W,
                         unsigned short* __restrict__ Wt, int K, int N)
{
  __shared__ float tile[32][33];
  size_t lofs = (size_t)blockIdx.z * K * N;
  const float* Wl = W + lofs;
  unsigned short* Wtl = Wt + lofs;
  int n0 = blockIdx.x * 32, k0 = blockIdx.y * 32;
  int c = threadIdx.x & 31, rb = threadIdx.x >> 5;
#pragma unroll
  for (int j = 0; j < 4; ++j) {
    int r = rb + j * 8;
    tile[r][c] = Wl[(size_t)(k0 + r) * N + (n0 + c)];
  }
  __syncthreads();
#pragma unroll
  for (int j = 0; j < 4; ++j) {
    int r = rb + j * 8;
    Wtl[(size_t)(n0 + r) * K + (k0 + c)] = f2bf(tile[c][r]);
  }
}

// In-place softmax over rows of width SEQ; also writes bf16 copy for AV GEMM.
__global__ __launch_bounds__(256)
void softmax_rows(float* __restrict__ p, unsigned short* __restrict__ pb)
{
  __shared__ float red[256];
  float* row = p + (size_t)blockIdx.x * SEQ;
  unsigned short* rowb = pb + (size_t)blockIdx.x * SEQ;
  int t = threadIdx.x;
  float a = row[t], b = row[t + 256];
  red[t] = fmaxf(a, b);
  __syncthreads();
  for (int s = 128; s > 0; s >>= 1) { if (t < s) red[t] = fmaxf(red[t], red[t + s]); __syncthreads(); }
  float mx = red[0];
  __syncthreads();
  float e0 = __expf(a - mx), e1 = __expf(b - mx);
  red[t] = e0 + e1;
  __syncthreads();
  for (int s = 128; s > 0; s >>= 1) { if (t < s) red[t] += red[t + s]; __syncthreads(); }
  float inv = 1.0f / red[0];
  float p0 = e0 * inv, p1 = e1 * inv;
  row[t] = p0;          row[t + 256] = p1;
  rowb[t] = f2bf(p0);   rowb[t + 256] = f2bf(p1);
}

// out = LayerNorm(a + b) (weight=1, bias=0, eps=1e-5); optional bf16 copy.
__global__ __launch_bounds__(256)
void add_ln(const float* __restrict__ a, const float* __restrict__ b,
            float* __restrict__ out, unsigned short* __restrict__ outb)
{
  __shared__ float red[256];
  size_t base = (size_t)blockIdx.x * D_MODEL;
  int t = threadIdx.x;
  float v[4]; float s = 0.f;
#pragma unroll
  for (int i = 0; i < 4; ++i) {
    int c = t + i * 256;
    v[i] = a[base + c] + b[base + c];
    s += v[i];
  }
  red[t] = s; __syncthreads();
  for (int st = 128; st > 0; st >>= 1) { if (t < st) red[t] += red[t + st]; __syncthreads(); }
  float mean = red[0] * (1.0f / D_MODEL);
  __syncthreads();
  float q = 0.f;
#pragma unroll
  for (int i = 0; i < 4; ++i) { float d = v[i] - mean; q += d * d; }
  red[t] = q; __syncthreads();
  for (int st = 128; st > 0; st >>= 1) { if (t < st) red[t] += red[t + st]; __syncthreads(); }
  float rstd = rsqrtf(red[0] * (1.0f / D_MODEL) + 1e-5f);
#pragma unroll
  for (int i = 0; i < 4; ++i) {
    int c = t + i * 256;
    float o = (v[i] - mean) * rstd;
    out[base + c] = o;
    if (outb) outb[base + c] = f2bf(o);
  }
}

// x = word_emb[tok] + pos_table[tok]; fp32 (residual) + bf16 (GEMM A operand)
__global__ __launch_bounds__(256)
void embed(const int* __restrict__ tok, const float* __restrict__ we,
           const float* __restrict__ pt, float* __restrict__ x,
           unsigned short* __restrict__ xb)
{
  int row = blockIdx.x;
  int tk = tok[row];
  size_t wb = (size_t)tk * D_MODEL;
  size_t ob = (size_t)row * D_MODEL;
#pragma unroll
  for (int i = 0; i < 4; ++i) {
    int c = threadIdx.x + i * 256;
    float val = we[wb + c] + pt[wb + c];
    x[ob + c] = val;
    xb[ob + c] = f2bf(val);
  }
}

extern "C" void kernel_launch(void* const* d_in, const int* in_sizes, int n_in,
                              void* d_out, int out_size, void* d_ws, size_t ws_size,
                              hipStream_t stream) {
  (void)in_sizes; (void)n_in; (void)out_size; (void)ws_size;

  const int*   enc = (const int*)d_in[0];
  const float* we  = (const float*)d_in[1];
  const float* pt  = (const float*)d_in[2];
  const float* Wq  = (const float*)d_in[3];
  const float* bq  = (const float*)d_in[4];
  const float* Wk  = (const float*)d_in[5];
  const float* bk  = (const float*)d_in[6];
  const float* Wv  = (const float*)d_in[7];
  const float* bv  = (const float*)d_in[8];
  const float* Wo  = (const float*)d_in[9];
  const float* bo  = (const float*)d_in[10];
  const float* W1  = (const float*)d_in[11];
  const float* b1  = (const float*)d_in[12];
  const float* W2  = (const float*)d_in[13];
  const float* b2  = (const float*)d_in[14];

  const size_t XE  = (size_t)ROWS * D_MODEL;              // 2,097,152
  const size_t FFE = (size_t)ROWS * DFF_;                 // 8,388,608
  const size_t ATE = (size_t)BATCH * NHEAD * SEQ * SEQ;   // 16,777,216
  const size_t WSM = (size_t)D_MODEL * D_MODEL;
  const size_t WLG = (size_t)D_MODEL * DFF_;

  // fp32 scratch
  float* x   = (float*)d_ws;
  float* x1  = x  + XE;
  float* tmp = x1 + XE;
  // bf16 scratch
  unsigned short* us     = (unsigned short*)(tmp + XE);
  unsigned short* x_bf   = us; us += XE;
  unsigned short* x1_bf  = us; us += XE;
  unsigned short* q_bf   = us; us += XE;
  unsigned short* k_bf   = us; us += XE;
  unsigned short* vT_bf  = us; us += XE;     // [B,H,DH,S]
  unsigned short* av_bf  = us; us += XE;
  unsigned short* ffh_bf = us; us += FFE;
  unsigned short* at_bf  = us; us += ATE;
  unsigned short* wqT = us; us += (size_t)NLAYER * WSM;
  unsigned short* wkT = us; us += (size_t)NLAYER * WSM;
  unsigned short* wvT = us; us += (size_t)NLAYER * WSM;
  unsigned short* woT = us; us += (size_t)NLAYER * WSM;
  unsigned short* w1T = us; us += (size_t)NLAYER * WLG;   // [DFF, D]
  unsigned short* w2T = us; us += (size_t)NLAYER * WLG;   // [D, DFF]

  float* outX  = (float*)d_out;                           // final hidden [B,S,D]
  float* corrs = outX + XE;                               // [L,B,H,S,S]

  // per-launch weight convert + transpose to N-major bf16
  conv_transpose_bf16<<<dim3(32, 32, NLAYER), 256, 0, stream>>>(Wq, wqT, D_MODEL, D_MODEL);
  conv_transpose_bf16<<<dim3(32, 32, NLAYER), 256, 0, stream>>>(Wk, wkT, D_MODEL, D_MODEL);
  conv_transpose_bf16<<<dim3(32, 32, NLAYER), 256, 0, stream>>>(Wv, wvT, D_MODEL, D_MODEL);
  conv_transpose_bf16<<<dim3(32, 32, NLAYER), 256, 0, stream>>>(Wo, woT, D_MODEL, D_MODEL);
  conv_transpose_bf16<<<dim3(128, 32, NLAYER), 256, 0, stream>>>(W1, w1T, D_MODEL, DFF_);
  conv_transpose_bf16<<<dim3(32, 128, NLAYER), 256, 0, stream>>>(W2, w2T, DFF_, D_MODEL);

  embed<<<ROWS, 256, 0, stream>>>(enc, we, pt, x, x_bf);

  const float inv_sqrt_dk = 0.125f;                       // 1/sqrt(64)
  const dim3 gProj(D_MODEL / BN, ROWS / BM, 1);           // (16,16,1)
  const dim3 gFF1(DFF_ / BN, ROWS / BM, 1);               // (64,16,1)
  const dim3 gScores(SEQ / BN, SEQ / BM, BATCH * NHEAD);  // (8,4,64)
  const dim3 gAV(DHEAD / BN, SEQ / BM, BATCH * NHEAD);    // (1,4,64)

  for (int l = 0; l < NLAYER; ++l) {
    const unsigned short* wq = wqT + (size_t)l * WSM;
    const unsigned short* wk = wkT + (size_t)l * WSM;
    const unsigned short* wv = wvT + (size_t)l * WSM;
    const unsigned short* wo = woT + (size_t)l * WSM;
    const unsigned short* w1 = w1T + (size_t)l * WLG;
    const unsigned short* w2 = w2T + (size_t)l * WLG;
    const float* bql = bq + (size_t)l * D_MODEL;
    const float* bkl = bk + (size_t)l * D_MODEL;
    const float* bvl = bv + (size_t)l * D_MODEL;
    const float* bol = bo + (size_t)l * D_MODEL;
    const float* b1l = b1 + (size_t)l * DFF_;
    const float* b2l = b2 + (size_t)l * D_MODEL;
    float* sc = corrs + (size_t)l * ATE;

    // Q, K projections -> bf16 row-major; V -> bf16 transposed [b,h,d,s]
    wmma_gemm_bf16<0><<<gProj, GT, 0, stream>>>(x_bf, D_MODEL, wq, D_MODEL,
        nullptr, 0, q_bf, D_MODEL, 0, bql, 0, 0.f, nullptr, D_MODEL);
    wmma_gemm_bf16<0><<<gProj, GT, 0, stream>>>(x_bf, D_MODEL, wk, D_MODEL,
        nullptr, 0, k_bf, D_MODEL, 0, bkl, 0, 0.f, nullptr, D_MODEL);
    wmma_gemm_bf16<0><<<gProj, GT, 0, stream>>>(x_bf, D_MODEL, wv, D_MODEL,
        nullptr, 0, vT_bf, 0, 1, bvl, 0, 0.f, nullptr, D_MODEL);

    // scores = scale * Q K^T (+ key-pad mask) -> corrs (fp32 output region)
    wmma_gemm_bf16<1><<<gScores, GT, 0, stream>>>(q_bf, D_MODEL, k_bf, D_MODEL,
        sc, SEQ, nullptr, 0, 0, nullptr, 0, inv_sqrt_dk, enc, DHEAD);

    // softmax in place on corrs; bf16 copy feeds the AV GEMM
    softmax_rows<<<BATCH * NHEAD * SEQ, 256, 0, stream>>>(sc, at_bf);

    // AV -> av_bf (bf16 row-major [B*S, H*DH])
    wmma_gemm_bf16<2><<<gAV, GT, 0, stream>>>(at_bf, SEQ, vT_bf, SEQ,
        nullptr, 0, av_bf, D_MODEL, 0, nullptr, 0, 0.f, nullptr, SEQ);

    // output projection (fp32 for residual) + residual + LN
    wmma_gemm_bf16<0><<<gProj, GT, 0, stream>>>(av_bf, D_MODEL, wo, D_MODEL,
        tmp, D_MODEL, nullptr, 0, 0, bol, 0, 0.f, nullptr, D_MODEL);
    add_ln<<<ROWS, 256, 0, stream>>>(x, tmp, x1, x1_bf);

    // FFN
    wmma_gemm_bf16<0><<<gFF1, GT, 0, stream>>>(x1_bf, D_MODEL, w1, D_MODEL,
        nullptr, 0, ffh_bf, DFF_, 0, b1l, 1, 0.f, nullptr, D_MODEL);
    wmma_gemm_bf16<0><<<gProj, GT, 0, stream>>>(ffh_bf, DFF_, w2, DFF_,
        tmp, D_MODEL, nullptr, 0, 0, b2l, 0, 0.f, nullptr, DFF_);
    add_ln<<<ROWS, 256, 0, stream>>>(x1, tmp,
                                     (l == NLAYER - 1) ? outX : x,
                                     (l == NLAYER - 1) ? nullptr : x_bf);
  }
}
